// CoAttention_15504831939249
// MI455X (gfx1250) — matrix operand order repaired and checked by reading, not compiled, not run on hardware
//
#include <hip/hip_runtime.h>
#include <hip/hip_bf16.h>
#include <cstdint>

#define BB 8
#define LL 2048
#define DD 1024

typedef __attribute__((ext_vector_type(16))) __bf16 v16bf;
typedef __attribute__((ext_vector_type(8)))  float  v8f;
typedef __attribute__((ext_vector_type(8)))  unsigned short u16x8;

union Frag {
  v16bf v;
  u16x8 h[2];
  unsigned short u[16];
  unsigned p[8];
};

// one f32 -> bf16 (round-half-up): 2 VALU
__device__ __forceinline__ unsigned short f2bf1(float f) {
  return (unsigned short)((__float_as_uint(f) + 0x8000u) >> 16);
}

// two f32 -> packed bf16x2 via v_perm_b32: 3 VALU per 2 elements.
// result: low 16 = bf16(a), high 16 = bf16(b)
__device__ __forceinline__ unsigned f2bf2(float a, float b) {
  unsigned ua = __float_as_uint(a) + 0x8000u;
  unsigned ub = __float_as_uint(b) + 0x8000u;
  return __builtin_amdgcn_perm(ub, ua, 0x07060302u);
}

__device__ __forceinline__ v8f wmma_bf16(const Frag& a, const Frag& b, v8f c) {
  return __builtin_amdgcn_wmma_f32_16x16x32_bf16(false, a.v, false, b.v,
                                                 (short)0, c, false, false);
}

// online softmax merge of (m,s) pairs
__device__ __forceinline__ void smerge(float& m, float& s, float m2, float s2) {
  float nm = fmaxf(m, m2);
  s = s * __expf(m - nm) + s2 * __expf(m2 - nm);
  m = nm;
}

// ---------------------------------------------------------------------------
// Projection GEMM: out[n,a] = sum_d X[n,d] * W[a,d], cast to bf16.
// WG = 8 waves, tile 128(M) x 64(N), K-chunks of 32. W^T staged via LDS.
// TRANSPOSED==1 stores out as [b][a][l] (for kT, so attention score
// B-fragments become contiguous loads); else row-major [b*L+l][a].
// ---------------------------------------------------------------------------
template <int TRANSPOSED>
__global__ __launch_bounds__(256, 4)
void proj_gemm(const float* __restrict__ X, const float* __restrict__ W,
               unsigned short* __restrict__ out)
{
  __shared__ __align__(16) unsigned short Bt[32][64];   // [k][n] bf16
  const int wave = threadIdx.x >> 5;
  const int lane = threadIdx.x & 31;
  const int mlow = lane & 15;
  const int up   = lane >> 4;
  const int khalf = up ? 8 : 0;
  const int klane = mlow + (up ? 16 : 0);
  const int tileM = blockIdx.x * 128;
  const int tileN = blockIdx.y * 64;
  const int rowM = tileM + wave * 16 + mlow;

  v8f acc[4] = {};
  #pragma unroll 1
  for (int k0 = 0; k0 < DD; k0 += 32) {
    __syncthreads();
    // stage W^T tile (32 k x 64 n) into LDS as bf16, two elements per thread
    for (int i = threadIdx.x; i < 64 * 16; i += 256) {
      int a = i >> 4, d2 = (i & 15) * 2;
      const float* wp = W + (size_t)(tileN + a) * DD + k0 + d2;
      unsigned pr = f2bf2(wp[0], wp[1]);
      Bt[d2][a]     = (unsigned short)(pr & 0xFFFFu);
      Bt[d2 + 1][a] = (unsigned short)(pr >> 16);
    }
    __syncthreads();
    // A fragment: lane = M row; halves carry K = {khalf..khalf+7, 16+khalf..}
    Frag af;
    const float* xr = X + (size_t)rowM * DD + k0 + khalf;
    #pragma unroll
    for (int j = 0; j < 4; ++j) af.p[j]     = f2bf2(xr[2 * j],      xr[2 * j + 1]);
    #pragma unroll
    for (int j = 0; j < 4; ++j) af.p[4 + j] = f2bf2(xr[16 + 2 * j], xr[17 + 2 * j]);
    #pragma unroll
    for (int nt = 0; nt < 4; ++nt) {
      Frag bfr;                       // B: lane = N, elements = K (table order)
      #pragma unroll
      for (int s = 0; s < 16; ++s) bfr.u[s] = Bt[khalf + (s & 7) + ((s >> 3) << 4)][nt * 16 + mlow];
      acc[nt] = wmma_bf16(af, bfr, acc[nt]);
    }
  }
  // C layout: VGPR i -> M = i (+8 for lanes>=16), N = lane&15
  const int mbase = tileM + wave * 16 + (up ? 8 : 0);
  #pragma unroll
  for (int nt = 0; nt < 4; ++nt) {
    #pragma unroll
    for (int i = 0; i < 8; ++i) {
      int m = mbase + i;
      int n = tileN + nt * 16 + mlow;
      unsigned short bv = f2bf1(acc[nt][i]);
      if (TRANSPOSED) {
        int b = m >> 11, l = m & (LL - 1);
        out[((size_t)b * DD + n) * LL + l] = bv;
      } else {
        out[(size_t)m * DD + n] = bv;
      }
    }
  }
}

// ---------------------------------------------------------------------------
// Pass 1: per-row softmax stats (max, sumexp) for self and other streams.
// Each wave owns 16 q rows; 4 key tiles (64 keys) are accumulated per D-sweep
// so each q A-fragment load feeds 8 WMMAs (all independent).
// ---------------------------------------------------------------------------
__global__ __launch_bounds__(256, 4)
void attn_stats(const unsigned short* __restrict__ qs,
                const unsigned short* __restrict__ qo,
                const unsigned short* __restrict__ kT,
                const float* __restrict__ mask,
                float* __restrict__ mS, float* __restrict__ sS,
                float* __restrict__ mO, float* __restrict__ sO)
{
  const int wave = threadIdx.x >> 5, lane = threadIdx.x & 31;
  const int mlow = lane & 15, up = lane >> 4;
  const int khalf = up ? 8 : 0;
  const int dlane = mlow + (up ? 16 : 0);
  const int gq = blockIdx.x * 128 + wave * 16;
  const int b = gq >> 11, q0 = gq & (LL - 1);
  const unsigned short* qsr = qs + ((size_t)gq + mlow) * DD;
  const unsigned short* qor = qo + ((size_t)gq + mlow) * DD;
  const float scale = 0.03125f;   // 1/sqrt(1024)

  float mSe[8], sSe[8], mOt[8], sOt[8];
  #pragma unroll
  for (int i = 0; i < 8; ++i) { mSe[i] = mOt[i] = -3.4e38f; sSe[i] = sOt[i] = 0.f; }

  #pragma unroll 1
  for (int kt = 0; kt < LL; kt += 64) {       // 4 key tiles of 16 per sweep
    v8f cs[4] = {}, co[4] = {};
    #pragma unroll 1
    for (int d0 = 0; d0 < DD; d0 += 32) {
      Frag aS, aO;
      aS.h[0] = *(const u16x8*)(qsr + d0 + khalf);
      aS.h[1] = *(const u16x8*)(qsr + d0 + khalf + 16);
      aO.h[0] = *(const u16x8*)(qor + d0 + khalf);
      aO.h[1] = *(const u16x8*)(qor + d0 + khalf + 16);
      const unsigned short* pk = kT + ((size_t)b * DD + d0) * LL + kt + mlow;
      #pragma unroll
      for (int t = 0; t < 4; ++t) {
        Frag bK;
        #pragma unroll
        for (int s = 0; s < 16; ++s) bK.u[s] = pk[(size_t)(khalf + (s & 7) + ((s >> 3) << 4)) * LL + t * 16];
        cs[t] = wmma_bf16(aS, bK, cs[t]);
        co[t] = wmma_bf16(aO, bK, co[t]);
      }
    }
    #pragma unroll
    for (int t = 0; t < 4; ++t) {
      float pen = (1.0f - mask[b * LL + kt + t * 16 + mlow]) * -100000.0f;
      #pragma unroll
      for (int i = 0; i < 8; ++i) {
        smerge(mSe[i], sSe[i], cs[t][i] * scale + pen, 1.0f);
        smerge(mOt[i], sOt[i], co[t][i] * scale + pen, 1.0f);
      }
    }
  }
  // merge partials across the 16 lanes sharing each row (xor < 16 stays in half)
  #pragma unroll
  for (int off = 1; off < 16; off <<= 1) {
    #pragma unroll
    for (int i = 0; i < 8; ++i) {
      smerge(mSe[i], sSe[i], __shfl_xor(mSe[i], off, 32), __shfl_xor(sSe[i], off, 32));
      smerge(mOt[i], sOt[i], __shfl_xor(mOt[i], off, 32), __shfl_xor(sOt[i], off, 32));
    }
  }
  if (mlow == 0) {
    int rbase = b * LL + q0 + (up ? 8 : 0);
    #pragma unroll
    for (int i = 0; i < 8; ++i) {
      mS[rbase + i] = mSe[i]; sS[rbase + i] = sSe[i];
      mO[rbase + i] = mOt[i]; sO[rbase + i] = sOt[i];
    }
  }
}

// ---------------------------------------------------------------------------
// Pass 2: recompute scores, normalize with stats, write p_self, and do the
// PV GEMMs. WG = one 16-row q tile; wave w computes score subtile
// (keys w>>1, stream w&1) into LDS, then all waves accumulate their own
// 128-column slice of the 1024-wide output.
// Score phase uses 2 independent accumulators (D split) and PV uses one
// accumulator set per stream so consecutive WMMAs are never RAW-dependent.
// ---------------------------------------------------------------------------
__global__ __launch_bounds__(256, 4)
void attn_main(const unsigned short* __restrict__ qs,
               const unsigned short* __restrict__ qo,
               const unsigned short* __restrict__ kT,
               const unsigned short* __restrict__ vv,
               const float* __restrict__ mask,
               const float* __restrict__ mS, const float* __restrict__ sS,
               const float* __restrict__ mO, const float* __restrict__ sO,
               float* __restrict__ out)
{
  __shared__ __align__(16) unsigned short pbuf[2][16][64]; // [stream][q][key] bf16
  const int wave = threadIdx.x >> 5, lane = threadIdx.x & 31;
  const int mlow = lane & 15, up = lane >> 4;
  const int khalf = up ? 8 : 0;
  const int dlane = mlow + (up ? 16 : 0);
  const int gq = blockIdx.x * 16;
  const int b = gq >> 11, q0 = gq & (LL - 1);
  const int st = wave & 1;       // score stream: 0=self, 1=other
  const int tsub = wave >> 1;    // key sub-tile 0..3
  const int col0 = wave * 128;   // this wave's output column slice
  const float scale = 0.03125f;

  const unsigned short* qrow = (st ? qo : qs) + ((size_t)gq + mlow) * DD;
  const float* mArr = st ? mO : mS;
  const float* sArr = st ? sO : sS;
  float mrow[8], srinv[8];
  #pragma unroll
  for (int i = 0; i < 8; ++i) {
    int r = b * LL + q0 + (up ? 8 : 0) + i;
    mrow[i] = mArr[r];
    srinv[i] = 1.0f / sArr[r];
  }
  float* pOut = out + (size_t)BB * LL * DD;   // p_self region of d_out
  v8f accS[8] = {}, accO[8] = {};

  #pragma unroll 1
  for (int key0 = 0; key0 < LL; key0 += 64) {
    __syncthreads();                          // pbuf from prev iter consumed
    // ---- score tile (16 q x 16 keys), K = D swept with 2 indep chains ----
    const int ks = key0 + tsub * 16;
    v8f c0 = {}, c1 = {};
    #pragma unroll 1
    for (int d0 = 0; d0 < DD; d0 += 64) {
      Frag a0, b0, a1, b1;
      a0.h[0] = *(const u16x8*)(qrow + d0 + khalf);
      a0.h[1] = *(const u16x8*)(qrow + d0 + khalf + 16);
      a1.h[0] = *(const u16x8*)(qrow + d0 + 32 + khalf);
      a1.h[1] = *(const u16x8*)(qrow + d0 + 32 + khalf + 16);
      const unsigned short* pk = kT + ((size_t)b * DD + d0) * LL + ks + mlow;
      #pragma unroll
      for (int s = 0; s < 16; ++s) {
        size_t kk = (size_t)(khalf + (s & 7) + ((s >> 3) << 4));
        b0.u[s] = pk[kk * LL];
        b1.u[s] = pk[(kk + 32) * LL];
      }
      c0 = wmma_bf16(a0, b0, c0);
      c1 = wmma_bf16(a1, b1, c1);
    }
    float pen = (1.0f - mask[b * LL + ks + mlow]) * -100000.0f;
    #pragma unroll
    for (int i = 0; i < 8; ++i) {
      int m = (up ? 8 : 0) + i;
      float p = __expf((c0[i] + c1[i]) * scale + pen - mrow[i]) * srinv[i];
      pbuf[st][m][tsub * 16 + mlow] = f2bf1(p);
      if (st == 0)
        pOut[((size_t)(b * LL + q0 + m)) * LL + ks + mlow] = p;
    }
    __syncthreads();
    // ---- PV: accS += p_self @ V, accO += p_other @ V over 64 keys ----
    #pragma unroll
    for (int kc = 0; kc < 2; ++kc) {
      Frag pa0, pa1;   // A fragments of p (lane=M, halves=K pattern)
      {
        const unsigned short* pb0 = &pbuf[0][mlow][kc * 32 + khalf];
        pa0.h[0] = *(const u16x8*)(pb0);
        pa0.h[1] = *(const u16x8*)(pb0 + 16);
        const unsigned short* pb1 = &pbuf[1][mlow][kc * 32 + khalf];
        pa1.h[0] = *(const u16x8*)(pb1);
        pa1.h[1] = *(const u16x8*)(pb1 + 16);
      }
      const unsigned short* vrow =
          vv + ((size_t)(b * LL + key0 + kc * 32)) * DD + col0 + mlow;
      #pragma unroll
      for (int nt = 0; nt < 8; ++nt) {
        Frag bv;   // B: lane = col, elements = keys (table order)
        #pragma unroll
        for (int s = 0; s < 16; ++s) bv.u[s] = vrow[(size_t)(khalf + (s & 7) + ((s >> 3) << 4)) * DD + nt * 16];
        accS[nt] = wmma_bf16(pa0, bv, accS[nt]);
        accO[nt] = wmma_bf16(pa1, bv, accO[nt]);
      }
    }
  }
  #pragma unroll
  for (int nt = 0; nt < 8; ++nt) {
    #pragma unroll
    for (int i = 0; i < 8; ++i) {
      int m = (up ? 8 : 0) + i;
      out[((size_t)(b * LL + q0 + m)) * DD + col0 + nt * 16 + mlow] =
          accS[nt][i] + accO[nt][i];
    }
  }
}

// ---------------------------------------------------------------------------
extern "C" void kernel_launch(void* const* d_in, const int* in_sizes, int n_in,
                              void* d_out, int out_size, void* d_ws, size_t ws_size,
                              hipStream_t stream) {
  (void)in_sizes; (void)n_in; (void)out_size; (void)ws_size;
  const float* x   = (const float*)d_in[0];
  const float* qu  = (const float*)d_in[1];
  const float* va  = (const float*)d_in[2];
  const float* msk = (const float*)d_in[3];
  const float* Wk  = (const float*)d_in[4];
  const float* Wqs = (const float*)d_in[5];
  const float* Wqo = (const float*)d_in[6];
  const float* Wv  = (const float*)d_in[7];

  char* ws = (char*)d_ws;
  const size_t matBytes = (size_t)BB * LL * DD * sizeof(unsigned short); // 32 MB
  unsigned short* kT  = (unsigned short*)(ws);                    // [b][d][l]
  unsigned short* qsb = (unsigned short*)(ws + matBytes);         // [b*l][d]
  unsigned short* qob = (unsigned short*)(ws + 2 * matBytes);
  unsigned short* vb  = (unsigned short*)(ws + 3 * matBytes);
  float* stats = (float*)(ws + 4 * matBytes);
  const size_t nrow = (size_t)BB * LL;
  float* mS = stats;            float* sS = stats + nrow;
  float* mO = stats + 2 * nrow; float* sO = stats + 3 * nrow;

  dim3 pg(BB * LL / 128, DD / 64, 1);
  proj_gemm<1><<<pg, 256, 0, stream>>>(x,  Wk,  kT);
  proj_gemm<0><<<pg, 256, 0, stream>>>(x,  Wqs, qsb);
  proj_gemm<0><<<pg, 256, 0, stream>>>(qu, Wqo, qob);
  proj_gemm<0><<<pg, 256, 0, stream>>>(va, Wv,  vb);
  attn_stats<<<BB * LL / 128, 256, 0, stream>>>(qsb, qob, kT, msk, mS, sS, mO, sO);
  attn_main<<<BB * LL / 16, 256, 0, stream>>>(qsb, qob, kT, vb, msk,
                                              mS, sS, mO, sO, (float*)d_out);
}
